// TrajectoryAttention_79044578116234
// MI455X (gfx1250) — matrix operand order repaired and checked by reading, not compiled, not run on hardware
//
#include <hip/hip_runtime.h>

// ---------------------------------------------------------------------------
// Types for WMMA fragments (gfx1250, wave32)
// ---------------------------------------------------------------------------
typedef __attribute__((ext_vector_type(16))) __bf16 v16bf;
typedef __attribute__((ext_vector_type(8)))  float  v8f;
typedef __attribute__((ext_vector_type(4)))  int    v4i;

struct alignas(16) U4 { unsigned int x, y, z, w; };

union Frag {
    v16bf v;
    U4    q[2];
};

__device__ inline v8f wmma_bf16(const Frag& a, const Frag& b, v8f c) {
    return __builtin_amdgcn_wmma_f32_16x16x32_bf16(
        false, a.v, false, b.v, (short)0, c, false, false);
}

__device__ inline unsigned short f2bf(float f) {
    union { float f; unsigned u; } x; x.f = f;
    unsigned r = x.u + 0x7FFFu + ((x.u >> 16) & 1u);   // round to nearest even
    return (unsigned short)(r >> 16);
}
__device__ inline float bf2f(unsigned short b) {
    union { unsigned u; float f; } x; x.u = ((unsigned)b) << 16;
    return x.f;
}

// ---------------------------------------------------------------------------
// CDNA5 async global->LDS staging (ASYNCcnt path). Guarded so the fallback
// (plain b128 load + ds_store) compiles when the builtin is absent
// (including the host pass, where AMDGCN builtins are not declared).
// Signature (probe-confirmed by diagnostic): (v4i AS1*, v4i AS3*, Ii, Ii).
// ---------------------------------------------------------------------------
#ifndef __has_builtin
#define __has_builtin(x) 0
#endif
#if __has_builtin(__builtin_amdgcn_global_load_async_to_lds_b128)
#define HAVE_ASYNC_LDS 1
#else
#define HAVE_ASYNC_LDS 0
#endif

#if HAVE_ASYNC_LDS
typedef __attribute__((address_space(1))) v4i gv4i;
typedef __attribute__((address_space(3))) v4i lv4i;
__device__ __forceinline__ void async_copy_b128(const void* g, void* l) {
    // LDS offset = low 32 bits of the flat address (ISA aperture layout)
    __builtin_amdgcn_global_load_async_to_lds_b128(
        (gv4i*)(unsigned long long)g,
        (lv4i*)(unsigned)(unsigned long long)l, 0, 0);
}
__device__ __forceinline__ void wait_async0() {
#if __has_builtin(__builtin_amdgcn_s_wait_asynccnt)
    __builtin_amdgcn_s_wait_asynccnt(0);
#else
    asm volatile("s_wait_asynccnt 0" ::: "memory");
#endif
}
#endif

// Problem constants
#define BB   2
#define NN   1569
#define CC   768
#define HH   12
#define DD   64
#define FF   8
#define PP   196
#define SS   1568
#define PPAD 224        // P padded to multiple of 32
#define SPAD 208        // P padded to multiple of 16 (score LDS stride)

// ---------------------------------------------------------------------------
// Generic bf16 GEMM:  C[MxN] = A[MxK] * Bt^T (+bias), fp32 accumulate.
// Bt is the weight PRE-TRANSPOSED to (N x K) so both A and B staging are
// pure b128 global->LDS copies (async when available, no sub-dword scatter).
// Workgroup: 256 threads = 8 waves, tile 128x64, each wave computes 32x32.
// B fragments: lane L holds column L%16, K = (L/16)*8 ... +8 and 16+(L/16)*8 ... +8.
// ---------------------------------------------------------------------------
__global__ __launch_bounds__(256) void gemm_bf16_wmma(
    const unsigned short* __restrict__ A, const unsigned short* __restrict__ Bt,
    float* Cf, unsigned short* Cb, const float* bias, int M, int N, int K)
{
    __shared__ alignas(16) unsigned short lA[128][32];   // 8 KB
    __shared__ alignas(16) unsigned short lBT[64][32];   // 4 KB

    const int tid   = threadIdx.x;
    const int lane  = tid & 31;
    const int wid   = tid >> 5;
    const int waveM = wid >> 1, waveN = wid & 1;
    const int tileM = blockIdx.y * 128;
    const int tileN = blockIdx.x * 64;

    const int rsel  = lane & 15;
    const int abase = (lane >> 4) << 3;     // A chunk base: 0 or 8
    const int bbase = (lane >> 4) << 3;     // B chunk base: 0 or 8

    // per-thread staging coordinates
    const int ar0 = (tid * 8) >> 5,  ac = (tid * 8) & 31;       // A: 2 chunks
    const int ar1 = ((tid + 256) * 8) >> 5;
    const int bn  = tid >> 2,        bc = (tid & 3) * 8;        // B: 1 chunk

    v8f acc[2][2] = {};

    for (int k0 = 0; k0 < K; k0 += 32) {
#if HAVE_ASYNC_LDS
        // async staging: copies bypass VGPRs; out-of-range rows are skipped
        // (their stale LDS rows only feed accumulator rows never stored)
        if (tileM + ar0 < M)
            async_copy_b128(A + (size_t)(tileM + ar0) * K + k0 + ac, &lA[ar0][ac]);
        if (tileM + ar1 < M)
            async_copy_b128(A + (size_t)(tileM + ar1) * K + k0 + ac, &lA[ar1][ac]);
        async_copy_b128(Bt + (size_t)(tileN + bn) * K + k0 + bc, &lBT[bn][bc]);
#else
        // stage A tile 128x32: two b128 copies per thread
        {
            U4 v0; v0.x = v0.y = v0.z = v0.w = 0u;
            U4 v1 = v0;
            if (tileM + ar0 < M) v0 = *(const U4*)(A + (size_t)(tileM + ar0) * K + k0 + ac);
            if (tileM + ar1 < M) v1 = *(const U4*)(A + (size_t)(tileM + ar1) * K + k0 + ac);
            *(U4*)&lA[ar0][ac] = v0;
            *(U4*)&lA[ar1][ac] = v1;
        }
        // stage B tile (already N-major): one b128 copy per thread
        {
            U4 val = *(const U4*)(Bt + (size_t)(tileN + bn) * K + k0 + bc);
            *(U4*)&lBT[bn][bc] = val;
        }
#endif
        // prefetch next K-tile into L2 while this one computes
        if (k0 + 32 < K) {
            if (tileM + ar0 < M)
                __builtin_prefetch(A + (size_t)(tileM + ar0) * K + k0 + 32 + ac, 0, 1);
            __builtin_prefetch(Bt + (size_t)(tileN + bn) * K + k0 + 32 + bc, 0, 1);
        }
#if HAVE_ASYNC_LDS
        wait_async0();
#endif
        __syncthreads();

        Frag a[2], b[2];
        #pragma unroll
        for (int i = 0; i < 2; ++i) {
            int r = waveM * 32 + i * 16 + rsel;
            a[i].q[0] = *(const U4*)&lA[r][abase];
            a[i].q[1] = *(const U4*)&lA[r][abase + 16];
        }
        #pragma unroll
        for (int j = 0; j < 2; ++j) {
            int n = waveN * 32 + j * 16 + rsel;
            b[j].q[0] = *(const U4*)&lBT[n][bbase];
            b[j].q[1] = *(const U4*)&lBT[n][bbase + 16];
        }
        #pragma unroll
        for (int i = 0; i < 2; ++i)
            #pragma unroll
            for (int j = 0; j < 2; ++j)
                acc[i][j] = wmma_bf16(a[i], b[j], acc[i][j]);
        __syncthreads();
    }

    const int mofs = (lane >= 16) ? 8 : 0;
    #pragma unroll
    for (int i = 0; i < 2; ++i)
        #pragma unroll
        for (int j = 0; j < 2; ++j) {
            int col = tileN + waveN * 32 + j * 16 + rsel;
            #pragma unroll
            for (int r = 0; r < 8; ++r) {
                int row = tileM + waveM * 32 + i * 16 + mofs + r;
                if (row < M) {
                    float v = acc[i][j][r];
                    if (bias) v += bias[col];
                    if (Cf) Cf[(size_t)row * N + col] = v;
                    else    Cb[(size_t)row * N + col] = f2bf(v);
                }
            }
        }
}

// ---------------------------------------------------------------------------
// Space attention: per (bh, frame f, 16-row Q tile), one wave.
//   scores = scale * Q K_f^T  (16 x 196) via WMMA, exact softmax over P,
//   O = P V_f (16 x 64) via WMMA, with V fragments read directly from the
//   padded transposed global layout Vt (B*h, F, d, PPAD) -- two 16B chunks
//   per lane, rows 448B aligned, L2-resident across the 98 Q tiles.
// Q/K layout: (B*h, N, d) bf16, token 0 is CLS (skipped here).
// ---------------------------------------------------------------------------
__global__ __launch_bounds__(32) void space_attn_wmma(
    const unsigned short* __restrict__ Q, const unsigned short* __restrict__ Km,
    const unsigned short* __restrict__ Vt,
    unsigned short* __restrict__ traj, unsigned short* __restrict__ xdiag,
    float scale)
{
    __shared__ alignas(16) float          sc[16][SPAD];   // 13.0 KB
    __shared__ alignas(16) unsigned short pr[16][PPAD];   //  7.0 KB

    const int lane = threadIdx.x;
    const int qt = blockIdx.x, f = blockIdx.y, bh = blockIdx.z;
    const int b = bh / HH, hh = bh % HH;
    const size_t headBase = (size_t)bh * NN * DD;
    const size_t vtBase   = ((size_t)bh * FF + f) * DD * PPAD;

    const int rsel  = lane & 15;
    const int abase = (lane >> 4) << 3;
    const int bbase = (lane >> 4) << 3;

    // Q fragments (row = s-tile row, K = d split in two 32-chunks)
    Frag a0, a1;
    {
        const unsigned short* qrow = Q + headBase + (size_t)(1 + qt * 16 + rsel) * DD;
        a0.q[0] = *(const U4*)(qrow + abase);
        a0.q[1] = *(const U4*)(qrow + abase + 16);
        a1.q[0] = *(const U4*)(qrow + 32 + abase);
        a1.q[1] = *(const U4*)(qrow + 32 + abase + 16);
    }

    // scores: 13 key tiles of 16 (tail keys masked out before softmax)
    for (int t = 0; t < 13; ++t) {
        int n = t * 16 + rsel;
        Frag b0, b1;
        if (n < PP) {
            const unsigned short* krow = Km + headBase + (size_t)(1 + f * PP + n) * DD;
            b0.q[0] = *(const U4*)(krow + bbase);
            b0.q[1] = *(const U4*)(krow + bbase + 16);
            b1.q[0] = *(const U4*)(krow + 32 + bbase);
            b1.q[1] = *(const U4*)(krow + 32 + bbase + 16);
        } else {
            b0.q[0].x = b0.q[0].y = b0.q[0].z = b0.q[0].w = 0u;
            b0.q[1] = b0.q[0]; b1.q[0] = b0.q[0]; b1.q[1] = b0.q[0];
        }
        v8f c = {};
        c = wmma_bf16(a0, b0, c);
        c = wmma_bf16(a1, b1, c);
        int mofs = (lane >= 16) ? 8 : 0;
        #pragma unroll
        for (int r = 0; r < 8; ++r) sc[mofs + r][t * 16 + rsel] = c[r];
    }
    __syncthreads();

    // exact softmax over the 196 valid keys, probs -> bf16 (zero padded)
    if (lane < 16) {
        float m = -1e30f;
        for (int j = 0; j < PP; ++j) m = fmaxf(m, sc[lane][j] * scale);
        float s = 0.f;
        for (int j = 0; j < PP; ++j) {
            float e = __expf(sc[lane][j] * scale - m);
            sc[lane][j] = e; s += e;
        }
        float inv = 1.0f / s;
        for (int j = 0; j < PP; ++j)    pr[lane][j] = f2bf(sc[lane][j] * inv);
        for (int j = PP; j < PPAD; ++j) pr[lane][j] = 0;
    }
    __syncthreads();

    // O = P @ V_f : K loop over 224 in steps of 32, N = d in 4 tiles of 16
    v8f o[4] = {};
    for (int kk = 0; kk < PPAD; kk += 32) {
        Frag ap;
        ap.q[0] = *(const U4*)&pr[rsel][kk + abase];
        ap.q[1] = *(const U4*)&pr[rsel][kk + abase + 16];
        #pragma unroll
        for (int j = 0; j < 4; ++j) {
            const unsigned short* vrow = Vt + vtBase + (size_t)(j * 16 + rsel) * PPAD + kk;
            Frag bv;
            bv.q[0] = *(const U4*)(vrow + bbase);
            bv.q[1] = *(const U4*)(vrow + bbase + 16);
            o[j] = wmma_bf16(ap, bv, o[j]);
        }
    }

    // store traj (B,S,F,C) and x_diag (B,S,C) for rows whose frame == f
    const int mofs = (lane >= 16) ? 8 : 0;
    #pragma unroll
    for (int j = 0; j < 4; ++j) {
        int dd = hh * DD + j * 16 + rsel;
        #pragma unroll
        for (int r = 0; r < 8; ++r) {
            int s = qt * 16 + mofs + r;
            unsigned short v = f2bf(o[j][r]);
            traj[(((size_t)(b * SS + s)) * FF + f) * CC + dd] = v;
            if (s / PP == f)
                xdiag[((size_t)(b * SS + s)) * CC + dd] = v;
        }
    }
}

// ---------------------------------------------------------------------------
// CLS attention: one block per (b,h); q[token0] vs all 1569 keys.
// Writes row 0 of the concat buffer (bf16).
// ---------------------------------------------------------------------------
__global__ __launch_bounds__(256) void cls_attn_kernel(
    const unsigned short* __restrict__ Q, const unsigned short* __restrict__ Km,
    const unsigned short* __restrict__ Vm,
    unsigned short* __restrict__ concatOut, float scale)
{
    __shared__ float logits[NN];
    __shared__ float red[256];
    __shared__ float qs[DD];
    const int tid = threadIdx.x, bh = blockIdx.x;
    const int b = bh / HH, hh = bh % HH;
    const size_t headBase = (size_t)bh * NN * DD;

    if (tid < DD) qs[tid] = bf2f(Q[headBase + tid]);
    __syncthreads();

    float lmax = -1e30f;
    for (int j = tid; j < NN; j += 256) {
        float dot = 0.f;
        for (int c = 0; c < DD; ++c) dot += qs[c] * bf2f(Km[headBase + (size_t)j * DD + c]);
        dot *= scale;
        logits[j] = dot;
        lmax = fmaxf(lmax, dot);
    }
    red[tid] = lmax; __syncthreads();
    for (int s = 128; s; s >>= 1) { if (tid < s) red[tid] = fmaxf(red[tid], red[tid + s]); __syncthreads(); }
    float m = red[0]; __syncthreads();

    float lsum = 0.f;
    for (int j = tid; j < NN; j += 256) { float e = __expf(logits[j] - m); logits[j] = e; lsum += e; }
    red[tid] = lsum; __syncthreads();
    for (int s = 128; s; s >>= 1) { if (tid < s) red[tid] += red[tid + s]; __syncthreads(); }
    float inv = 1.0f / red[0]; __syncthreads();

    if (tid < DD) {
        float o = 0.f;
        for (int j = 0; j < NN; ++j) o += logits[j] * bf2f(Vm[headBase + (size_t)j * DD + tid]);
        concatOut[(size_t)b * NN * CC + hh * DD + tid] = f2bf(o * inv);
    }
}

// ---------------------------------------------------------------------------
// Temporal attention over F=8 frames per (b,s,h). Also emits the `attn`
// output (B,h,S,F) fp32. Output rows go to concat buffer rows 1..S (bf16).
// ---------------------------------------------------------------------------
__global__ __launch_bounds__(96) void temporal_attn_kernel(
    const unsigned short* __restrict__ q2, const unsigned short* __restrict__ k2,
    const unsigned short* __restrict__ traj,
    unsigned short* __restrict__ concatOut, float* __restrict__ attnOut,
    float scale)
{
    __shared__ float att[HH][FF];
    const int t = threadIdx.x;
    const int idx = blockIdx.x;            // b*S + s
    const int b = idx / SS, s = idx % SS;
    const int hh = t >> 3, f = t & 7;

    const size_t qoff = (size_t)idx * CC + hh * DD;
    const size_t koff = ((size_t)idx * FF + f) * CC + hh * DD;
    float dot = 0.f;
    for (int c = 0; c < DD; ++c) dot += bf2f(q2[qoff + c]) * bf2f(k2[koff + c]);
    dot *= scale;

    float m = dot;
    for (int o = 4; o; o >>= 1) m = fmaxf(m, __shfl_xor(m, o, 8));
    float e = __expf(dot - m);
    float sm = e;
    for (int o = 4; o; o >>= 1) sm += __shfl_xor(sm, o, 8);
    float a = e / sm;

    att[hh][f] = a;
    attnOut[(((size_t)(b * HH + hh)) * SS + s) * FF + f] = a;
    __syncthreads();

    const int hh2 = t >> 3;
    const int dd0 = (t & 7) * 8;
    const size_t trow = (size_t)idx * FF;
    #pragma unroll
    for (int i = 0; i < 8; ++i) {
        int dd = dd0 + i;
        float o = 0.f;
        for (int ff = 0; ff < FF; ++ff)
            o += att[hh2][ff] * bf2f(traj[(trow + ff) * CC + hh2 * DD + dd]);
        concatOut[((size_t)b * NN + 1 + s) * CC + hh2 * DD + dd] = f2bf(o);
    }
}

// ---------------------------------------------------------------------------
// Conversion / packing helpers
// ---------------------------------------------------------------------------
__global__ void f32_to_bf16_kernel(const float* __restrict__ src,
                                   unsigned short* __restrict__ dst, size_t n)
{
    size_t i = (size_t)blockIdx.x * 256 + threadIdx.x;
    if (i < n) dst[i] = f2bf(src[i]);
}

// dst (N x K) = transpose of src (K x N, leading dim ldsrc), fp32 -> bf16
__global__ void f32_to_bf16_transpose_kernel(const float* __restrict__ src,
                                             unsigned short* __restrict__ dst,
                                             int K, int N, int ldsrc)
{
    size_t i = (size_t)blockIdx.x * 256 + threadIdx.x;
    if (i < (size_t)N * K) {
        int n = (int)(i / K), k = (int)(i % K);
        dst[i] = f2bf(src[(size_t)k * ldsrc + n]);
    }
}

__global__ void zero_u16_kernel(unsigned short* __restrict__ p, size_t n)
{
    size_t i = (size_t)blockIdx.x * 256 + threadIdx.x;
    if (i < n) p[i] = 0;
}

// qkv (B,N,3C) bf16 -> q/k/v head-major (B*h, N, d) bf16, plus V in the
// padded transposed per-frame layout Vt (B*h, F, d, PPAD).
__global__ void pack_qkv_kernel(const unsigned short* __restrict__ qkv,
                                unsigned short* __restrict__ q,
                                unsigned short* __restrict__ k,
                                unsigned short* __restrict__ v,
                                unsigned short* __restrict__ vt)
{
    size_t i = (size_t)blockIdx.x * 256 + threadIdx.x;
    const size_t total = (size_t)BB * NN * CC;
    if (i >= total) return;
    size_t bn = i / CC; int c = (int)(i % CC);
    int b = (int)(bn / NN), n = (int)(bn % NN);
    int hh = c / DD, dd = c % DD;
    int bh = b * HH + hh;
    size_t dstIdx = ((size_t)bh * NN + n) * DD + dd;
    size_t srcRow = bn * (3 * CC);
    unsigned short vv = qkv[srcRow + 2 * CC + c];
    q[dstIdx] = qkv[srcRow + c];
    k[dstIdx] = qkv[srcRow + CC + c];
    v[dstIdx] = vv;
    if (n >= 1) {
        int s = n - 1, f = s / PP, p = s % PP;
        vt[(((size_t)bh * FF + f) * DD + dd) * PPAD + p] = vv;
    }
}

// ---------------------------------------------------------------------------
// Host orchestration
// ---------------------------------------------------------------------------
extern "C" void kernel_launch(void* const* d_in, const int* in_sizes, int n_in,
                              void* d_out, int out_size, void* d_ws, size_t ws_size,
                              hipStream_t stream)
{
    (void)in_sizes; (void)n_in; (void)out_size; (void)ws_size;
    const float* x      = (const float*)d_in[0];
    const float* W_qkv  = (const float*)d_in[1];
    const float* W_pq   = (const float*)d_in[2];
    const float* W_pkv  = (const float*)d_in[3];
    const float* W_proj = (const float*)d_in[4];
    const float* b_proj = (const float*)d_in[5];

    float* out     = (float*)d_out;                          // (B,1569,C) fp32
    float* attnOut = out + (size_t)BB * NN * CC;             // (B,h,S,F)  fp32

    const float scale = 0.125f;   // d^-0.5, d=64

    char* ws = (char*)d_ws;
    size_t cur = 0;
    auto alloc_u16 = [&](size_t n) -> unsigned short* {
        unsigned short* p = (unsigned short*)(ws + cur);
        cur += ((n * 2 + 255) / 256) * 256;
        return p;
    };

    const size_t MROWS = (size_t)BB * NN;    // 3138
    unsigned short* x_bf     = alloc_u16(MROWS * CC);
    unsigned short* wqkvT_bf = alloc_u16((size_t)3 * CC * CC);   // (3C x C)
    unsigned short* wpqT_bf  = alloc_u16((size_t)CC * CC);       // (C x C)
    unsigned short* wpkvkT_bf= alloc_u16((size_t)CC * CC);       // k-half, (C x C)
    unsigned short* wprojT_bf= alloc_u16((size_t)CC * CC);       // (C x C)
    unsigned short* qkv_bf   = alloc_u16(MROWS * 3 * CC);
    unsigned short* q_bf     = alloc_u16((size_t)BB * HH * NN * DD);
    unsigned short* k_bf     = alloc_u16((size_t)BB * HH * NN * DD);
    unsigned short* v_bf     = alloc_u16((size_t)BB * HH * NN * DD);
    unsigned short* vt_bf    = alloc_u16((size_t)BB * HH * FF * DD * PPAD);
    unsigned short* traj_bf  = alloc_u16((size_t)BB * SS * FF * CC);
    unsigned short* xdiag_bf = alloc_u16((size_t)BB * SS * CC);
    unsigned short* q2_bf    = alloc_u16((size_t)BB * SS * CC);
    unsigned short* k2_bf    = alloc_u16((size_t)BB * SS * FF * CC);
    unsigned short* concat_bf= alloc_u16(MROWS * CC);

    auto cdiv = [](size_t a, size_t b) { return (unsigned)((a + b - 1) / b); };

    // 1) fp32 -> bf16 conversions (weights transposed to N-major)
    {
        size_t n = MROWS * CC;
        f32_to_bf16_kernel<<<cdiv(n, 256), 256, 0, stream>>>(x, x_bf, n);
        n = (size_t)3 * CC * CC;
        f32_to_bf16_transpose_kernel<<<cdiv(n, 256), 256, 0, stream>>>(
            W_qkv, wqkvT_bf, CC, 3 * CC, 3 * CC);
        n = (size_t)CC * CC;
        f32_to_bf16_transpose_kernel<<<cdiv(n, 256), 256, 0, stream>>>(
            W_pq, wpqT_bf, CC, CC, CC);
        f32_to_bf16_transpose_kernel<<<cdiv(n, 256), 256, 0, stream>>>(
            W_proj, wprojT_bf, CC, CC, CC);
        f32_to_bf16_transpose_kernel<<<cdiv(n, 256), 256, 0, stream>>>(
            W_pkv, wpkvkT_bf, CC, CC, 2 * CC);   // k-half only; v2 is dead code
        size_t nv = (size_t)BB * HH * FF * DD * PPAD;
        zero_u16_kernel<<<cdiv(nv, 256), 256, 0, stream>>>(vt_bf, nv);
    }

    // 2) qkv = x @ W_qkv   (3138 x 768 x 2304) -> bf16
    {
        dim3 g(3 * CC / 64, cdiv(MROWS, 128));
        gemm_bf16_wmma<<<g, 256, 0, stream>>>(x_bf, wqkvT_bf, nullptr, qkv_bf,
                                              nullptr, (int)MROWS, 3 * CC, CC);
    }

    // 3) split into head-major q/k/v (+ padded transposed V)
    {
        size_t n = MROWS * CC;
        pack_qkv_kernel<<<cdiv(n, 256), 256, 0, stream>>>(qkv_bf, q_bf, k_bf,
                                                          v_bf, vt_bf);
    }

    // 4) CLS attention -> concat row 0
    cls_attn_kernel<<<BB * HH, 256, 0, stream>>>(q_bf, k_bf, v_bf, concat_bf, scale);

    // 5) space attention per (bh, f, q-tile) -> traj + x_diag
    {
        dim3 g(SS / 16, FF, BB * HH);
        space_attn_wmma<<<g, 32, 0, stream>>>(q_bf, k_bf, vt_bf, traj_bf,
                                              xdiag_bf, scale);
    }

    // 6) q2 = x_diag @ W_pq ; k2 = traj @ W_pkv[:, :C]
    {
        dim3 g2(CC / 64, cdiv((size_t)BB * SS, 128));
        gemm_bf16_wmma<<<g2, 256, 0, stream>>>(xdiag_bf, wpqT_bf, nullptr, q2_bf,
                                               nullptr, BB * SS, CC, CC);
        dim3 g3(CC / 64, cdiv((size_t)BB * SS * FF, 128));
        gemm_bf16_wmma<<<g3, 256, 0, stream>>>(traj_bf, wpkvkT_bf, nullptr, k2_bf,
                                               nullptr, BB * SS * FF, CC, CC);
    }

    // 7) temporal attention -> concat rows 1..S, attn output
    temporal_attn_kernel<<<BB * SS, 96, 0, stream>>>(q2_bf, k2_bf, traj_bf,
                                                     concat_bf, attnOut, scale);

    // 8) out = concat @ W_proj + b_proj  -> fp32 d_out
    {
        dim3 g(CC / 64, cdiv(MROWS, 128));
        gemm_bf16_wmma<<<g, 256, 0, stream>>>(concat_bf, wprojT_bf, out, nullptr,
                                              b_proj, (int)MROWS, CC, CC);
    }
}
